// GATBlock_6150393168639
// MI455X (gfx1250) — matrix operand order repaired and checked by reading, not compiled, not run on hardware
//
#include <hip/hip_runtime.h>
#include <hip/hip_bf16.h>

typedef __attribute__((ext_vector_type(16))) _Float16 v16h;
typedef __attribute__((ext_vector_type(8)))  _Float16 v8h;
typedef __attribute__((ext_vector_type(8)))  float    v8f;
typedef __attribute__((ext_vector_type(4)))  float    v4f;

#define BN_EPS     1e-5f
#define NEG_SLOPE  0.2f

static constexpr int Bc  = 32;            // batch
static constexpr int Cin = 32;            // input channels
static constexpr int Tc  = 2048;          // sequence length
static constexpr int Fc  = 32;            // feature dim per head
static constexpr int Kc  = 9;             // conv kernel
static constexpr int Nn  = Bc * Tc;       // 65536 nodes
static constexpr long NCT = (long)Bc * Cin * Tc;  // 2,097,152 elements

// ---------------------------------------------------------------- utilities

__global__ void fill_u32_kernel(unsigned* __restrict__ p, unsigned v, long n) {
    long id = (long)blockIdx.x * blockDim.x + threadIdx.x;
    if (id < n) p[id] = v;
}

__global__ void f32_to_f16_kernel(const float* __restrict__ in, _Float16* __restrict__ out, int n) {
    int id = blockIdx.x * blockDim.x + threadIdx.x;
    if (id < n) out[id] = (_Float16)in[id];
}

// monotone float <-> uint key for atomic max over signed floats
__device__ __forceinline__ unsigned f2key(float f) {
    unsigned b = __float_as_uint(f);
    return (b & 0x80000000u) ? ~b : (b | 0x80000000u);
}
__device__ __forceinline__ float key2f(unsigned k) {
    unsigned b = (k & 0x80000000u) ? (k ^ 0x80000000u) : ~k;
    return __uint_as_float(b);
}

// ---------------------------------------------------------------- BatchNorm

// bn3d: stats over (B,T) per channel; one block per channel c
__global__ void bn3d_stats_kernel(const float* __restrict__ x,
                                  float* __restrict__ mean, float* __restrict__ rstd) {
    __shared__ float ssum[256], ssq[256];
    int c = blockIdx.x;
    float s = 0.f, q = 0.f;
    for (int i = threadIdx.x; i < Bc * Tc; i += blockDim.x) {
        int b = i / Tc, t = i % Tc;
        float v = x[(long)b * Cin * Tc + (long)c * Tc + t];
        s += v; q += v * v;
    }
    ssum[threadIdx.x] = s; ssq[threadIdx.x] = q;
    __syncthreads();
    for (int st = 128; st > 0; st >>= 1) {
        if ((int)threadIdx.x < st) {
            ssum[threadIdx.x] += ssum[threadIdx.x + st];
            ssq[threadIdx.x]  += ssq[threadIdx.x + st];
        }
        __syncthreads();
    }
    if (threadIdx.x == 0) {
        float inv = 1.f / (float)(Bc * Tc);
        float m   = ssum[0] * inv;
        float var = ssq[0] * inv - m * m;
        mean[c] = m;
        rstd[c] = rsqrtf(var + BN_EPS);
    }
}

// apply bn3d and emit both f32 (for conv) and f16 (for GEMM A-matrix view xn)
__global__ void bn3d_apply_kernel(const float* __restrict__ x,
                                  const float* __restrict__ mean, const float* __restrict__ rstd,
                                  const float* __restrict__ g, const float* __restrict__ be,
                                  float* __restrict__ xb, _Float16* __restrict__ xh) {
    long id = (long)blockIdx.x * blockDim.x + threadIdx.x;
    if (id >= NCT) return;
    int c = (int)((id / Tc) % Cin);
    float v = (x[id] - mean[c]) * rstd[c] * g[c] + be[c];
    xb[id] = v;
    xh[id] = (_Float16)v;
}

// bn2d: stats over N per channel; one block per channel (bias cancels -> omitted)
__global__ void bn2d_stats_kernel(const float* __restrict__ h, int C,
                                  float* __restrict__ mean, float* __restrict__ rstd) {
    __shared__ float ssum[256], ssq[256];
    int c = blockIdx.x;
    float s = 0.f, q = 0.f;
    for (int n = threadIdx.x; n < Nn; n += blockDim.x) {
        float v = h[(long)n * C + c];
        s += v; q += v * v;
    }
    ssum[threadIdx.x] = s; ssq[threadIdx.x] = q;
    __syncthreads();
    for (int st = 128; st > 0; st >>= 1) {
        if ((int)threadIdx.x < st) {
            ssum[threadIdx.x] += ssum[threadIdx.x + st];
            ssq[threadIdx.x]  += ssq[threadIdx.x + st];
        }
        __syncthreads();
    }
    if (threadIdx.x == 0) {
        float inv = 1.f / (float)Nn;
        float m   = ssum[0] * inv;
        float var = ssq[0] * inv - m * m;
        mean[c] = m;
        rstd[c] = rsqrtf(var + BN_EPS);
    }
}

__global__ void bn2d_apply_kernel(const float* __restrict__ h,
                                  const float* __restrict__ mean, const float* __restrict__ rstd,
                                  const float* __restrict__ g, const float* __restrict__ be,
                                  int C, float* __restrict__ fOut, _Float16* __restrict__ hOut) {
    long id = (long)blockIdx.x * blockDim.x + threadIdx.x;
    if (id >= (long)Nn * C) return;
    int c = (int)(id % C);
    float y = (h[id] - mean[c]) * rstd[c] * g[c] + be[c];
    y = y > 0.f ? y : 0.f;
    if (fOut) fOut[id] = y;
    if (hOut) hOut[id] = (_Float16)y;
}

// ---------------------------------------------------------------- conv1d skip path

__global__ void conv1d_relu_kernel(const float* __restrict__ xb, const float* __restrict__ W,
                                   const float* __restrict__ bias, float* __restrict__ res) {
    long id = (long)blockIdx.x * blockDim.x + threadIdx.x;
    if (id >= NCT) return;                       // B*F*T == NCT (F == Cin)
    int t  = (int)(id % Tc);
    int co = (int)((id / Tc) % Fc);
    int b  = (int)(id / ((long)Tc * Fc));
    float s = bias[co];
    for (int ci = 0; ci < Cin; ++ci) {
        const float* xr = xb + (long)b * Cin * Tc + (long)ci * Tc;
        const float* wr = W + (long)co * Cin * Kc + ci * Kc;
#pragma unroll
        for (int k = 0; k < Kc; ++k) {
            int tt = t + k - Kc / 2;
            float xv = (tt >= 0 && tt < Tc) ? xr[tt] : 0.f;
            s += xv * wr[k];
        }
    }
    res[id] = s > 0.f ? s : 0.f;
}

// ---------------------------------------------------------------- WMMA GEMM
// C[M,NOUT] = A[M,K](f16) * W[NOUT,K]^T (f16), f32 accumulate.
// One wave32 per 16x16 tile; A and B both use contiguous k order (k = 16*hl + i).
template<int K, int NOUT>
__global__ void gemm_wmma_kernel(const _Float16* __restrict__ A, const _Float16* __restrict__ Bw,
                                 float* __restrict__ C, int M) {
    int wave = threadIdx.x >> 5;
    int lane = threadIdx.x & 31;
    int tile = blockIdx.x * (blockDim.x >> 5) + wave;
    constexpr int NT = NOUT / 16;
    if (tile >= (M / 16) * NT) return;
    int mT = tile / NT, nT = tile % NT;
    int hl = lane >> 4, m16 = lane & 15;

    const _Float16* arow = A  + (long)(mT * 16 + m16) * K;   // row m of A
    const _Float16* brow = Bw + (long)(nT * 16 + m16) * K;   // W row n = B column n

    v8f acc = {};
#pragma unroll
    for (int kt = 0; kt < K / 32; ++kt) {
        // A 16x32 f16: elem i holds K = kt*32 + hl*16 + i
        union { v16h v; v8h h[2]; } av;
        av.v = *(const v16h*)(arow + kt * 32 + hl * 16);
        // B 32x16 f16: elem i holds K = kt*32 + hl*16 + i of column m16
        v16h bv = *(const v16h*)(brow + kt * 32 + hl * 16);
        acc = __builtin_amdgcn_wmma_f32_16x16x32_f16(false, av.v, false, bv,
                                                     (short)0, acc, false, false);
    }
    // D: VGPR r -> row 8*hl + r, col m16
    float* crow = C + (long)(mT * 16 + 8 * hl) * NOUT + nT * 16 + m16;
#pragma unroll
    for (int r = 0; r < 8; ++r)
        crow[(long)r * NOUT] = acc[r];
}

// ---------------------------------------------------------------- GAT edge phase

__device__ __forceinline__ void edge_endpoints(const int* __restrict__ ei, int E, int e,
                                               int& src, int& dst) {
    if (e < E) { src = ei[e]; dst = ei[E + e]; }
    else       { src = e - E; dst = e - E; }        // self loops appended
}

// score[e,h] = sum_f leakyrelu(xl[src,h,f] + xr[dst,h,f]) * att[h,f]   (F = 32)
__global__ void gat_score_kernel(const float* __restrict__ xl, const float* __restrict__ xr,
                                 const float* __restrict__ att, const int* __restrict__ ei,
                                 int E, int E2, int H, float* __restrict__ score) {
    long tid = (long)blockIdx.x * blockDim.x + threadIdx.x;
    if (tid >= (long)E2 * H) return;
    int e = (int)(tid / H), h = (int)(tid % H);
    int src, dst; edge_endpoints(ei, E, e, src, dst);
    int HF = H * 32;
    const float* pl = xl + (long)src * HF + h * 32;
    const float* pr = xr + (long)dst * HF + h * 32;
    const float* pa = att + h * 32;
    __builtin_prefetch(pl, 0, 0);
    __builtin_prefetch(pr, 0, 0);
    float s = 0.f;
#pragma unroll
    for (int f = 0; f < 32; f += 4) {
        v4f a = *(const v4f*)(pl + f);
        v4f b = *(const v4f*)(pr + f);
        v4f t = *(const v4f*)(pa + f);
#pragma unroll
        for (int j = 0; j < 4; ++j) {
            float v = a[j] + b[j];
            v = v > 0.f ? v : NEG_SLOPE * v;
            s += v * t[j];
        }
    }
    score[tid] = s;
}

__global__ void seg_max_kernel(const float* __restrict__ score, const int* __restrict__ ei,
                               int E, int E2, int H, unsigned* __restrict__ mx) {
    long tid = (long)blockIdx.x * blockDim.x + threadIdx.x;
    if (tid >= (long)E2 * H) return;
    int e = (int)(tid / H), h = (int)(tid % H);
    int src, dst; edge_endpoints(ei, E, e, src, dst);
    atomicMax(&mx[(long)dst * H + h], f2key(score[tid]));
}

__global__ void seg_expsum_kernel(float* __restrict__ score, const int* __restrict__ ei,
                                  int E, int E2, int H, const unsigned* __restrict__ mx,
                                  float* __restrict__ den) {
    long tid = (long)blockIdx.x * blockDim.x + threadIdx.x;
    if (tid >= (long)E2 * H) return;
    int e = (int)(tid / H), h = (int)(tid % H);
    int src, dst; edge_endpoints(ei, E, e, src, dst);
    float m  = key2f(mx[(long)dst * H + h]);
    float ex = expf(score[tid] - m);
    score[tid] = ex;                              // score buffer now holds exp()
    atomicAdd(&den[(long)dst * H + h], ex);
}

// out[dst, h, f] += xl[src, h, f] * (ex[e,h] / den[dst,h])
__global__ void aggregate_kernel(const float* __restrict__ xl, const float* __restrict__ ex,
                                 const float* __restrict__ den, const int* __restrict__ ei,
                                 int E, int E2, int H, float* __restrict__ out) {
    int HF = H * 32;
    long tid = (long)blockIdx.x * blockDim.x + threadIdx.x;
    if (tid >= (long)E2 * HF) return;
    int e = (int)(tid / HF), j = (int)(tid % HF);
    int h = j >> 5;
    int src, dst; edge_endpoints(ei, E, e, src, dst);
    float alpha = ex[(long)e * H + h] / den[(long)dst * H + h];
    atomicAdd(&out[(long)dst * HF + j], xl[(long)src * HF + j] * alpha);
}

// ---------------------------------------------------------------- epilogue

__global__ void final_add_kernel(const float* __restrict__ res, const float* __restrict__ h2,
                                 float* __restrict__ out) {
    long id = (long)blockIdx.x * blockDim.x + threadIdx.x;
    if (id < NCT) out[id] = res[id] + h2[id];
}

// ---------------------------------------------------------------- launch

extern "C" void kernel_launch(void* const* d_in, const int* in_sizes, int n_in,
                              void* d_out, int out_size, void* d_ws, size_t ws_size,
                              hipStream_t stream) {
    const float* x      = (const float*)d_in[0];
    const int*   ei     = (const int*)  d_in[1];
    const float* g0     = (const float*)d_in[3];
    const float* be0    = (const float*)d_in[4];
    const float* Wskip  = (const float*)d_in[5];
    const float* bskip  = (const float*)d_in[6];
    const float* Wl1    = (const float*)d_in[7];
    const float* Wr1    = (const float*)d_in[8];
    const float* att1   = (const float*)d_in[9];
    const float* g1     = (const float*)d_in[11];
    const float* be1    = (const float*)d_in[12];
    const float* Wl2    = (const float*)d_in[13];
    const float* Wr2    = (const float*)d_in[14];
    const float* att2   = (const float*)d_in[15];
    const float* g2     = (const float*)d_in[17];
    const float* be2    = (const float*)d_in[18];
    float* out = (float*)d_out;

    const int E  = in_sizes[1] / 2;
    const int E2 = E + Nn;

    // bump allocator over d_ws
    char* ws = (char*)d_ws;
    size_t off = 0;
    auto alloc = [&](size_t bytes) -> void* {
        off = (off + 255) & ~(size_t)255;
        void* p = ws + off;
        off += bytes;
        return p;
    };

    float*     mean0 = (float*)alloc(Cin * 4);
    float*     rstd0 = (float*)alloc(Cin * 4);
    float*     xb    = (float*)alloc(NCT * 4);
    _Float16*  xh    = (_Float16*)alloc(NCT * 2);
    float*     resd  = (float*)alloc(NCT * 4);
    _Float16*  Wl1h  = (_Float16*)alloc(64 * 32 * 2);
    _Float16*  Wr1h  = (_Float16*)alloc(64 * 32 * 2);
    _Float16*  Wl2h  = (_Float16*)alloc(32 * 64 * 2);
    _Float16*  Wr2h  = (_Float16*)alloc(32 * 64 * 2);
    float*     xl1   = (float*)alloc((long)Nn * 64 * 4);
    float*     xr1   = (float*)alloc((long)Nn * 64 * 4);
    float*     sc1   = (float*)alloc((long)E2 * 2 * 4);
    unsigned*  mx1   = (unsigned*)alloc((long)Nn * 2 * 4);
    float*     dn1   = (float*)alloc((long)Nn * 2 * 4);
    float*     out1  = (float*)alloc((long)Nn * 64 * 4);
    float*     mean1 = (float*)alloc(64 * 4);
    float*     rstd1 = (float*)alloc(64 * 4);
    _Float16*  h1h   = (_Float16*)alloc((long)Nn * 64 * 2);
    float*     xl2   = (float*)alloc((long)Nn * 32 * 4);
    float*     xr2   = (float*)alloc((long)Nn * 32 * 4);
    float*     sc2   = (float*)alloc((long)E2 * 4);
    unsigned*  mx2   = (unsigned*)alloc((long)Nn * 4);
    float*     dn2   = (float*)alloc((long)Nn * 4);
    float*     out2  = (float*)alloc((long)Nn * 32 * 4);
    float*     mean2 = (float*)alloc(32 * 4);
    float*     rstd2 = (float*)alloc(32 * 4);
    float*     h2    = (float*)alloc((long)Nn * 32 * 4);

    const int TPB = 256;
    auto blocks = [&](long n) { return (unsigned)((n + TPB - 1) / TPB); };

    // BN0 + f16 staging of xn view
    bn3d_stats_kernel<<<Cin, TPB, 0, stream>>>(x, mean0, rstd0);
    bn3d_apply_kernel<<<blocks(NCT), TPB, 0, stream>>>(x, mean0, rstd0, g0, be0, xb, xh);

    // residual skip path
    conv1d_relu_kernel<<<blocks(NCT), TPB, 0, stream>>>(xb, Wskip, bskip, resd);

    // weights -> f16
    f32_to_f16_kernel<<<blocks(2048), TPB, 0, stream>>>(Wl1, Wl1h, 2048);
    f32_to_f16_kernel<<<blocks(2048), TPB, 0, stream>>>(Wr1, Wr1h, 2048);
    f32_to_f16_kernel<<<blocks(2048), TPB, 0, stream>>>(Wl2, Wl2h, 2048);
    f32_to_f16_kernel<<<blocks(2048), TPB, 0, stream>>>(Wr2, Wr2h, 2048);

    // ---- GAT layer 1: K=32, NOUT=64 (H=2, F=32) ----
    {
        const int tiles = (Nn / 16) * (64 / 16);
        gemm_wmma_kernel<32, 64><<<tiles / 8, TPB, 0, stream>>>(xh, Wl1h, xl1, Nn);
        gemm_wmma_kernel<32, 64><<<tiles / 8, TPB, 0, stream>>>(xh, Wr1h, xr1, Nn);

        fill_u32_kernel<<<blocks((long)Nn * 2), TPB, 0, stream>>>(mx1, 0u, (long)Nn * 2);
        fill_u32_kernel<<<blocks((long)Nn * 2), TPB, 0, stream>>>((unsigned*)dn1, 0u, (long)Nn * 2);
        fill_u32_kernel<<<blocks((long)Nn * 64), TPB, 0, stream>>>((unsigned*)out1, 0u, (long)Nn * 64);

        gat_score_kernel<<<blocks((long)E2 * 2), TPB, 0, stream>>>(xl1, xr1, att1, ei, E, E2, 2, sc1);
        seg_max_kernel  <<<blocks((long)E2 * 2), TPB, 0, stream>>>(sc1, ei, E, E2, 2, mx1);
        seg_expsum_kernel<<<blocks((long)E2 * 2), TPB, 0, stream>>>(sc1, ei, E, E2, 2, mx1, dn1);
        aggregate_kernel<<<blocks((long)E2 * 64), TPB, 0, stream>>>(xl1, sc1, dn1, ei, E, E2, 2, out1);

        bn2d_stats_kernel<<<64, TPB, 0, stream>>>(out1, 64, mean1, rstd1);
        bn2d_apply_kernel<<<blocks((long)Nn * 64), TPB, 0, stream>>>(out1, mean1, rstd1, g1, be1,
                                                                     64, nullptr, h1h);
    }

    // ---- GAT layer 2: K=64, NOUT=32 (H=1, F=32) ----
    {
        const int tiles = (Nn / 16) * (32 / 16);
        gemm_wmma_kernel<64, 32><<<tiles / 8, TPB, 0, stream>>>(h1h, Wl2h, xl2, Nn);
        gemm_wmma_kernel<64, 32><<<tiles / 8, TPB, 0, stream>>>(h1h, Wr2h, xr2, Nn);

        fill_u32_kernel<<<blocks((long)Nn), TPB, 0, stream>>>(mx2, 0u, (long)Nn);
        fill_u32_kernel<<<blocks((long)Nn), TPB, 0, stream>>>((unsigned*)dn2, 0u, (long)Nn);
        fill_u32_kernel<<<blocks((long)Nn * 32), TPB, 0, stream>>>((unsigned*)out2, 0u, (long)Nn * 32);

        gat_score_kernel<<<blocks((long)E2), TPB, 0, stream>>>(xl2, xr2, att2, ei, E, E2, 1, sc2);
        seg_max_kernel  <<<blocks((long)E2), TPB, 0, stream>>>(sc2, ei, E, E2, 1, mx2);
        seg_expsum_kernel<<<blocks((long)E2), TPB, 0, stream>>>(sc2, ei, E, E2, 1, mx2, dn2);
        aggregate_kernel<<<blocks((long)E2 * 32), TPB, 0, stream>>>(xl2, sc2, dn2, ei, E, E2, 1, out2);

        bn2d_stats_kernel<<<32, TPB, 0, stream>>>(out2, 32, mean2, rstd2);
        bn2d_apply_kernel<<<blocks((long)Nn * 32), TPB, 0, stream>>>(out2, mean2, rstd2, g2, be2,
                                                                     32, h2, nullptr);
    }

    // out = residual + h (flat views coincide)
    final_add_kernel<<<blocks(NCT), TPB, 0, stream>>>(resd, h2, out);
}
